// NoisyTopKGating_47433618817511
// MI455X (gfx1250) — compile-verified, matrix operand order checked by reading, not run on hardware
//
#include <hip/hip_runtime.h>
#include <math.h>

typedef _Float16 v16h __attribute__((ext_vector_type(16)));
typedef float    v8f  __attribute__((ext_vector_type(8)));

#define N_TOK 16384
#define DIM   2048
#define NEXP  64
#define WAVES_PER_BLOCK 4      // each wave: 32 rows x 16 experts
#define ROWS_PER_BLOCK  32
#define LDS_STRIDE      65     // 64 + 1 pad -> conflict-free per-row scans

// d_out layout (flat float, reference return order):
//   [0,       32768)  combined_weights [N,2]
//   [32768,   65536)  top_k_indices    [N,2] (as float)
//   [65536, 1114112)  clean_logits     [N,64]
#define OUT_W_OFF 0
#define OUT_I_OFF (N_TOK * 2)
#define OUT_C_OFF (N_TOK * 4)

// Build one f16 16x32 fragment (A-style; B mirrored) from f32 memory.
// p = row_base + k0 + laneHi*8 : two contiguous 8-float runs (p, p+16).
__device__ __forceinline__ v16h cvt_frag(const float* __restrict__ p) {
    const float4 a0 = *(const float4*)(p);
    const float4 a1 = *(const float4*)(p + 4);
    const float4 b0 = *(const float4*)(p + 16);
    const float4 b1 = *(const float4*)(p + 20);
    v16h r;
    r[0]  = (_Float16)a0.x; r[1]  = (_Float16)a0.y; r[2]  = (_Float16)a0.z; r[3]  = (_Float16)a0.w;
    r[4]  = (_Float16)a1.x; r[5]  = (_Float16)a1.y; r[6]  = (_Float16)a1.z; r[7]  = (_Float16)a1.w;
    r[8]  = (_Float16)b0.x; r[9]  = (_Float16)b0.y; r[10] = (_Float16)b0.z; r[11] = (_Float16)b0.w;
    r[12] = (_Float16)b1.x; r[13] = (_Float16)b1.y; r[14] = (_Float16)b1.z; r[15] = (_Float16)b1.w;
    return r;
}

__global__ __launch_bounds__(WAVES_PER_BLOCK * 32)
void noisy_topk_gating_kernel(const float* __restrict__ x,
                              const float* __restrict__ Wg,
                              const float* __restrict__ Wn,
                              const float* __restrict__ noise,
                              float* __restrict__ out)
{
    __shared__ float lds[ROWS_PER_BLOCK * LDS_STRIDE];

    const int lane   = threadIdx.x & 31;
    const int wave   = threadIdx.x >> 5;      // expert quarter: experts [wave*16, wave*16+16)
    const int laneM  = lane & 15;             // row / expert within half-wave
    const int laneHi = lane >> 4;
    const int blockRow0 = blockIdx.x * ROWS_PER_BLOCK;

    // accumulators: [strip 0/1] x {gate, noise}
    v8f accG0 = (v8f)0.0f, accG1 = (v8f)0.0f;
    v8f accN0 = (v8f)0.0f, accN1 = (v8f)0.0f;

    const float* xA0 = x  + (size_t)(blockRow0 + laneM) * DIM       + laneHi * 8;
    const float* xA1 = xA0 + (size_t)16 * DIM;
    const float* wgp = Wg + (size_t)(wave * 16 + laneM) * DIM       + laneHi * 8;
    const float* wnp = Wn + (size_t)(wave * 16 + laneM) * DIM       + laneHi * 8;

    // ---- contraction: D=2048 in K=32 steps of f16 WMMA (f32 accumulate) ----
    #pragma unroll 2
    for (int k0 = 0; k0 < DIM; k0 += 32) {
        const v16h a0 = cvt_frag(xA0 + k0);
        const v16h a1 = cvt_frag(xA1 + k0);
        const v16h bg = cvt_frag(wgp + k0);
        const v16h bn = cvt_frag(wnp + k0);

        accG0 = __builtin_amdgcn_wmma_f32_16x16x32_f16(false, a0, false, bg, (short)0, accG0, false, false);
        accG1 = __builtin_amdgcn_wmma_f32_16x16x32_f16(false, a1, false, bg, (short)0, accG1, false, false);
        accN0 = __builtin_amdgcn_wmma_f32_16x16x32_f16(false, a0, false, bn, (short)0, accN0, false, false);
        accN1 = __builtin_amdgcn_wmma_f32_16x16x32_f16(false, a1, false, bn, (short)0, accN1, false, false);
    }

    // ---- epilogue: clean logits out, noisy logits into LDS ----
    // C/D layout: VGPR j -> (M=j lanes 0-15, M=j+8 lanes 16-31), N=laneM
    const int col = wave * 16 + laneM;
    #pragma unroll
    for (int s = 0; s < 2; ++s) {
        const v8f g = s ? accG1 : accG0;
        const v8f n = s ? accN1 : accN0;
        #pragma unroll
        for (int j = 0; j < 8; ++j) {
            const int lrow = s * 16 + j + laneHi * 8;     // row within block
            const int row  = blockRow0 + lrow;
            const float c  = g[j];
            out[OUT_C_OFF + (size_t)row * NEXP + col] = c;

            const float z  = n[j];
            const float sp = (z > 20.0f) ? z : log1pf(__expf(z));   // softplus
            const float nv = noise[(size_t)row * NEXP + col];
            lds[lrow * LDS_STRIDE + col] = c + sp * nv;
        }
    }
    __syncthreads();

    // ---- top-2 + softmax: wave 0, one lane per token row ----
    if (threadIdx.x < ROWS_PER_BLOCK) {
        const int myRow = blockRow0 + threadIdx.x;
        const float* rowp = lds + threadIdx.x * LDS_STRIDE;
        float m1 = -INFINITY, m2 = -INFINITY;
        int   i1 = 0,         i2 = 0;
        #pragma unroll 8
        for (int e = 0; e < NEXP; ++e) {
            const float v = rowp[e];
            if (v > m1)      { m2 = m1; i2 = i1; m1 = v; i1 = e; }
            else if (v > m2) { m2 = v; i2 = e; }
        }
        const float d  = __expf(m2 - m1);
        const float w1 = 1.0f / (1.0f + d);
        const float w2 = d * w1;
        out[OUT_W_OFF + (size_t)myRow * 2 + 0] = w1;
        out[OUT_W_OFF + (size_t)myRow * 2 + 1] = w2;
        out[OUT_I_OFF + (size_t)myRow * 2 + 0] = (float)i1;
        out[OUT_I_OFF + (size_t)myRow * 2 + 1] = (float)i2;
    }
}

extern "C" void kernel_launch(void* const* d_in, const int* in_sizes, int n_in,
                              void* d_out, int out_size, void* d_ws, size_t ws_size,
                              hipStream_t stream)
{
    const float* x     = (const float*)d_in[0];
    const float* Wg    = (const float*)d_in[1];
    const float* Wn    = (const float*)d_in[2];
    const float* noise = (const float*)d_in[3];
    float* out = (float*)d_out;

    dim3 grid(N_TOK / ROWS_PER_BLOCK);          // 512 blocks -> fills all WGPs
    dim3 block(WAVES_PER_BLOCK * 32);           // 128 threads = 4 waves
    noisy_topk_gating_kernel<<<grid, block, 0, stream>>>(x, Wg, Wn, noise, out);
}
